// Attention_73040213836409
// MI455X (gfx1250) — matrix operand order repaired and checked by reading, not compiled, not run on hardware
//
#include <hip/hip_runtime.h>
#include <hip/hip_bf16.h>
#include <stdint.h>

// ---------------------------------------------------------------------------
// Problem constants (match reference)
// ---------------------------------------------------------------------------
#define B_   4
#define S_   1024
#define DIM_ 2048
#define H_   32
#define KVH_ 8
#define D_   64
#define REP_ (H_ / KVH_)          // 4
#define SCALE_ 0.125f             // 1/sqrt(64)

// ---------------------------------------------------------------------------
// WMMA types / helpers (CDNA5 gfx1250, wave32)
// ---------------------------------------------------------------------------
typedef __attribute__((ext_vector_type(16))) __bf16 v16bf;
typedef __attribute__((ext_vector_type(8)))  float  v8f;

union BF16Frag { v16bf v; uint32_t u[8]; };
union BF16Pack { __bf16 h[2]; uint32_t u; };

__device__ __forceinline__ v8f wmma_bf16(v16bf a, v16bf b, v8f c) {
  // D = A(16x32 bf16) * B(32x16 bf16) + C(16x16 f32)
  return __builtin_amdgcn_wmma_f32_16x16x32_bf16(
      /*neg_a=*/false, a, /*neg_b=*/false, b,
      /*c_mod=*/(short)0, c, /*reuse_a=*/false, /*reuse_b=*/false);
}

// A-fragment: 16x32, row-major source (stride in elements, kbase = K offset).
// ISA 7.12.2: lane l (l<16) row M=l holds K 0..7 (v0..3) and 16..23 (v4..7);
// lane l+16 holds K 8..15 and 24..31.  K-pairs contiguous -> b32 reads.
__device__ __forceinline__ v16bf load_frag_a(const __bf16* base, int stride, int kbase) {
  const int lane  = threadIdx.x & 31;
  const int row   = lane & 15;
  const int khalf = lane >> 4;
  BF16Frag f;
  const uint32_t* p32 = (const uint32_t*)(base + (size_t)row * stride + kbase);
#pragma unroll
  for (int j = 0; j < 8; ++j) {
    const int kloc = ((j < 4) ? 0 : 16) + khalf * 8 + 2 * (j & 3);
    f.u[j] = p32[kloc >> 1];
  }
  return f.v;
}

// B-fragment: 32x16 (KxN), source stored [n][k] with k contiguous.
// Lane l: N = l&15, K = (j<4 ? 0 : 16) + (l>>4)*8 + 2(j&3)..+1 (same order as A).
__device__ __forceinline__ v16bf load_frag_b(const __bf16* base, int stride,
                                             int colbase, int kbase) {
  const int lane  = threadIdx.x & 31;
  const int col   = lane & 15;
  const int khalf = lane >> 4;
  BF16Frag f;
  const uint32_t* p32 =
      (const uint32_t*)(base + (size_t)(colbase + col) * stride + kbase);
#pragma unroll
  for (int j = 0; j < 8; ++j) f.u[j] = p32[((j < 4) ? 0 : 8) + khalf * 4 + (j & 3)];
  return f.v;
}

// ---------------------------------------------------------------------------
// CDNA5 async global->LDS copy (ASYNCcnt-tracked, bypasses VGPRs)
// ---------------------------------------------------------------------------
__device__ __forceinline__ uint32_t lds_off_u32(const void* p) {
  // Generic LDS address: low 32 bits are the LDS byte offset (ISA 10.2).
  return (uint32_t)(uintptr_t)p;
}

__device__ __forceinline__ void async_load_b128(uint32_t lds_byte_off, const void* gptr) {
  asm volatile("global_load_async_to_lds_b128 %0, %1, off"
               :
               : "v"(lds_byte_off), "v"((uint64_t)(uintptr_t)gptr)
               : "memory");
}

__device__ __forceinline__ void wait_async0() {
  asm volatile("s_wait_asynccnt 0" ::: "memory");
}

// ---------------------------------------------------------------------------
// f32 -> bf16 packed convert (2 elems / thread)
// ---------------------------------------------------------------------------
__global__ void f32_to_bf16x2_kernel(const float2* __restrict__ in,
                                     uint32_t* __restrict__ out, int n2) {
  int i = blockIdx.x * blockDim.x + threadIdx.x;
  if (i >= n2) return;
  const float2 t = in[i];
  BF16Pack p;
  p.h[0] = (__bf16)t.x;
  p.h[1] = (__bf16)t.y;
  out[i] = p.u;
}

// ---------------------------------------------------------------------------
// Tiled transpose: in f32 [K][N] -> out bf16 [N][K]   (grid: K/32 x N/32)
// ---------------------------------------------------------------------------
__global__ __launch_bounds__(128)
void transpose_f32_to_bf16_kernel(const float* __restrict__ in,
                                  __bf16* __restrict__ out, int K, int N) {
  __shared__ float t[32][33];
  const int k0 = blockIdx.x * 32, n0 = blockIdx.y * 32;
  const int tx = threadIdx.x & 31, ty = threadIdx.x >> 5;   // ty in 0..3
#pragma unroll
  for (int i = ty; i < 32; i += 4)
    t[i][tx] = in[(size_t)(k0 + i) * N + n0 + tx];
  __syncthreads();
#pragma unroll
  for (int i = ty; i < 32; i += 4)
    out[(size_t)(n0 + i) * K + k0 + tx] = (__bf16)t[tx][i];
}

// ---------------------------------------------------------------------------
// V transpose: vf f32 (B,S,KVH,D) -> vt bf16 (B,KVH,D,S)
// grid: (S/32, D/32, B*KVH)
// ---------------------------------------------------------------------------
__global__ __launch_bounds__(128)
void transpose_v_kernel(const float* __restrict__ in, __bf16* __restrict__ out) {
  __shared__ float t[32][33];
  const int s0 = blockIdx.x * 32, d0 = blockIdx.y * 32;
  const int bk = blockIdx.z;                         // b*KVH + kvh
  const int b = bk / KVH_, kvh = bk % KVH_;
  const int tx = threadIdx.x & 31, ty = threadIdx.x >> 5;
#pragma unroll
  for (int i = ty; i < 32; i += 4)
    t[i][tx] = in[(((size_t)b * S_ + s0 + i) * KVH_ + kvh) * D_ + d0 + tx];
  __syncthreads();
#pragma unroll
  for (int i = ty; i < 32; i += 4)
    out[((size_t)bk * D_ + d0 + i) * S_ + s0 + tx] = (__bf16)t[tx][i];
}

// ---------------------------------------------------------------------------
// RoPE (interleaved pairs) f32 -> bf16, packed dword stores
// layout: (B, S, nheads, D); freqs: (S, D/2)
// ---------------------------------------------------------------------------
__global__ void rope_to_bf16_kernel(const float2* __restrict__ in,
                                    const float* __restrict__ cosf,
                                    const float* __restrict__ sinf,
                                    uint32_t* __restrict__ out,
                                    int nheads, int total) {
  int idx = blockIdx.x * blockDim.x + threadIdx.x;
  if (idx >= total) return;
  const int d2   = idx & 31;                 // D/2 = 32
  const int rest = idx >> 5;
  const int h    = rest % nheads;
  const int bs   = rest / nheads;            // b*S + s
  const int s    = bs & (S_ - 1);
  const size_t p = (((size_t)bs * nheads) + h) * 32 + d2;   // pair index
  const float2 t = in[p];
  const float c = cosf[s * 32 + d2], sn = sinf[s * 32 + d2];
  BF16Pack pk;
  pk.h[0] = (__bf16)(t.x * c - t.y * sn);
  pk.h[1] = (__bf16)(t.x * sn + t.y * c);
  out[p] = pk.u;
}

// ---------------------------------------------------------------------------
// bf16 WMMA GEMM:  C[MxN] f32 = A[MxK] bf16 (row-major) * Wt[NxK] bf16 (N-major)
// Block tile 64x64, 4 waves, each 32x32 (2x2 WMMA tiles), K step 32.
// Double-buffered async global->LDS staging pipeline.
// ---------------------------------------------------------------------------
__global__ __launch_bounds__(128)
void gemm_bf16_kernel(const __bf16* __restrict__ A, const __bf16* __restrict__ Wt,
                      float* __restrict__ C, int M, int N, int K) {
  __shared__ __attribute__((aligned(16))) __bf16 As[2][64 * 32];   // [m][k]
  __shared__ __attribute__((aligned(16))) __bf16 Bs[2][64 * 32];   // [n][k]

  const int tid = threadIdx.x;
  const int wv  = tid >> 5;
  const int m0  = blockIdx.y * 64;
  const int n0  = blockIdx.x * 64;
  const int wm  = (wv >> 1) * 32;
  const int wn  = (wv & 1) * 32;

  const uint32_t as_base = lds_off_u32(As);
  const uint32_t bs_base = lds_off_u32(Bs);
  // Per-thread fixed staging slot: 4 async b128 transfers per tile pair.
  const int sm = tid >> 1, skv = (tid & 1) * 16;      // rows 0..63, k 0/16

  auto issue_tile = [&](int k0, int buf) {
    const uint32_t boff = (uint32_t)buf * (64 * 32 * 2);
#pragma unroll
    for (int it = 0; it < 2; ++it) {
      const int kv = skv + it * 8;
      async_load_b128(as_base + boff + (uint32_t)(sm * 32 + kv) * 2,
                      A + (size_t)(m0 + sm) * K + k0 + kv);
      async_load_b128(bs_base + boff + (uint32_t)(sm * 32 + kv) * 2,
                      Wt + (size_t)(n0 + sm) * K + k0 + kv);
    }
  };

  v8f acc[2][2] = {};

  const int T = K / 32;
  issue_tile(0, 0);
  for (int t = 0; t < T; ++t) {
    wait_async0();
    __syncthreads();
    if (t + 1 < T) issue_tile((t + 1) * 32, (t + 1) & 1);

    const __bf16* Ab = As[t & 1];
    const __bf16* Bb = Bs[t & 1];
    const v16bf a0 = load_frag_a(Ab + (size_t)(wm +  0) * 32, 32, 0);
    const v16bf a1 = load_frag_a(Ab + (size_t)(wm + 16) * 32, 32, 0);
    const v16bf b0 = load_frag_b(Bb, 32, wn +  0, 0);
    const v16bf b1 = load_frag_b(Bb, 32, wn + 16, 0);
    acc[0][0] = wmma_bf16(a0, b0, acc[0][0]);
    acc[0][1] = wmma_bf16(a0, b1, acc[0][1]);
    acc[1][0] = wmma_bf16(a1, b0, acc[1][0]);
    acc[1][1] = wmma_bf16(a1, b1, acc[1][1]);
  }

  const int lane  = tid & 31;
  const int coln  = lane & 15;
  const int rbase = (lane >> 4) * 8;
#pragma unroll
  for (int i = 0; i < 2; ++i)
#pragma unroll
    for (int j = 0; j < 2; ++j)
#pragma unroll
      for (int r = 0; r < 8; ++r) {
        const int row = m0 + wm + 16 * i + rbase + r;
        const int col = n0 + wn + 16 * j + coln;
        C[(size_t)row * N + col] = acc[i][j][r];
      }
}

// ---------------------------------------------------------------------------
// Causal GQA flash attention (bf16 in, bf16 out, f32 accumulators).
// Q: (B,S,H,D) bf16 ; K: (B,S,KVH,D) bf16 ; V: (B,KVH,D,S) bf16 (pre-transposed)
// Grid: (S/64, H, B); 128 threads (4 waves); wave owns 16 q rows.
// Double-buffered async KV staging pipeline.
// ---------------------------------------------------------------------------
__global__ __launch_bounds__(128)
void flash_attn_kernel(const __bf16* __restrict__ Q, const __bf16* __restrict__ Kc,
                       const __bf16* __restrict__ Vt_g, __bf16* __restrict__ O) {
  __shared__ __attribute__((aligned(16))) __bf16 Ks[2][64 * 64];   // [kv][d]
  __shared__ __attribute__((aligned(16))) __bf16 Vt[2][64 * 64];   // [d][kv]
  __shared__ __attribute__((aligned(16))) __bf16 Pw[4][16 * 64];   // per-wave P

  const int tid    = threadIdx.x;
  const int lane   = tid & 31;
  const int wv     = tid >> 5;
  const int qblock = blockIdx.x;
  const int h      = blockIdx.y;
  const int b      = blockIdx.z;
  const int kvh    = h / REP_;
  const int qr0    = qblock * 64 + wv * 16;

  const int coln  = lane & 15;
  const int rbase = (lane >> 4) * 8;

  const uint32_t ks_base = lds_off_u32(Ks);
  const uint32_t vt_base = lds_off_u32(Vt);

  // Q fragments held in registers for the whole kernel (row stride H*D).
  const __bf16* qbase = Q + (((size_t)b * S_ + qr0) * H_ + h) * D_;
  const v16bf aq0 = load_frag_a(qbase, H_ * D_, 0);
  const v16bf aq1 = load_frag_a(qbase, H_ * D_, 32);

  const __bf16* kg = Kc + ((size_t)b * S_ * KVH_ + kvh) * D_;       // + s*KVH_*D_
  const __bf16* vg = Vt_g + ((size_t)(b * KVH_ + kvh) * D_) * S_;   // + d*S_ + s

  // Per-thread fixed staging slot: 8 async b128 transfers per KV tile pair.
  const int sr = tid >> 1, sc = (tid & 1) * 32;       // rows 0..63, col 0/32

  auto issue_kv = [&](int kb0, int buf) {
    const uint32_t boff = (uint32_t)buf * (64 * 64 * 2);
#pragma unroll
    for (int it = 0; it < 4; ++it) {
      const int c8 = sc + it * 8;
      async_load_b128(ks_base + boff + (uint32_t)(sr * 64 + c8) * 2,
                      kg + (size_t)(kb0 + sr) * (KVH_ * D_) + c8);
      async_load_b128(vt_base + boff + (uint32_t)(sr * 64 + c8) * 2,
                      vg + (size_t)sr * S_ + kb0 + c8);
    }
  };

  v8f o[4] = {};
  float mrow[8], lrow[8];
#pragma unroll
  for (int r = 0; r < 8; ++r) { mrow[r] = -3.0e38f; lrow[r] = 0.0f; }

  issue_kv(0, 0);
  for (int kvb = 0; kvb <= qblock; ++kvb) {
    const int kb0 = kvb * 64;
    wait_async0();
    __syncthreads();
    if (kvb + 1 <= qblock) issue_kv((kvb + 1) * 64, (kvb + 1) & 1);

    const __bf16* Kb = Ks[kvb & 1];
    const __bf16* Vb = Vt[kvb & 1];

    // S = Q @ K^T  (contraction over D=64 -> two WMMA k-steps)
    v8f st[4] = {};
#pragma unroll
    for (int j = 0; j < 4; ++j) {
      st[j] = wmma_bf16(aq0, load_frag_b(Kb, 64, 16 * j, 0),  st[j]);
      st[j] = wmma_bf16(aq1, load_frag_b(Kb, 64, 16 * j, 32), st[j]);
    }

    // Scale + causal mask (diagonal block only).
    const bool diag = (kvb == qblock);
#pragma unroll
    for (int j = 0; j < 4; ++j)
#pragma unroll
      for (int r = 0; r < 8; ++r) {
        float v = st[j][r] * SCALE_;
        if (diag && (kb0 + 16 * j + coln > qr0 + rbase + r)) v = -1.0e30f;
        st[j][r] = v;
      }

    // Online softmax: row max / sum via 16-lane-group shuffles.
    float mnew[8];
#pragma unroll
    for (int r = 0; r < 8; ++r) {
      float m = st[0][r];
#pragma unroll
      for (int j = 1; j < 4; ++j) m = fmaxf(m, st[j][r]);
#pragma unroll
      for (int mask = 1; mask < 16; mask <<= 1)
        m = fmaxf(m, __shfl_xor(m, mask, 32));
      mnew[r] = fmaxf(m, mrow[r]);
    }
    float alpha[8];
#pragma unroll
    for (int r = 0; r < 8; ++r) { alpha[r] = __expf(mrow[r] - mnew[r]); mrow[r] = mnew[r]; }

    float rs[8];
#pragma unroll
    for (int r = 0; r < 8; ++r) rs[r] = 0.0f;
#pragma unroll
    for (int j = 0; j < 4; ++j)
#pragma unroll
      for (int r = 0; r < 8; ++r) {
        const float p = __expf(st[j][r] - mnew[r]);
        st[j][r] = p;
        rs[r] += p;
      }
#pragma unroll
    for (int r = 0; r < 8; ++r) {
#pragma unroll
      for (int mask = 1; mask < 16; mask <<= 1)
        rs[r] += __shfl_xor(rs[r], mask, 32);
      lrow[r] = lrow[r] * alpha[r] + rs[r];
    }
#pragma unroll
    for (int j = 0; j < 4; ++j)
#pragma unroll
      for (int r = 0; r < 8; ++r) o[j][r] *= alpha[r];

    // P (C-layout) -> LDS -> A-fragments.
#pragma unroll
    for (int j = 0; j < 4; ++j)
#pragma unroll
      for (int r = 0; r < 8; ++r)
        Pw[wv][(rbase + r) * 64 + 16 * j + coln] = (__bf16)st[j][r];
    __syncthreads();

    const v16bf ap0 = load_frag_a(Pw[wv], 64, 0);
    const v16bf ap1 = load_frag_a(Pw[wv], 64, 32);
#pragma unroll
    for (int j = 0; j < 4; ++j) {
      o[j] = wmma_bf16(ap0, load_frag_b(Vb, 64, 16 * j, 0),  o[j]);
      o[j] = wmma_bf16(ap1, load_frag_b(Vb, 64, 16 * j, 32), o[j]);
    }
  }

  // Epilogue: divide by row sums, store bf16 in (B,S,H,D) layout.
#pragma unroll
  for (int j = 0; j < 4; ++j)
#pragma unroll
    for (int r = 0; r < 8; ++r) {
      const float val = o[j][r] / lrow[r];
      const size_t idx =
          (((size_t)b * S_ + qr0 + rbase + r) * H_ + h) * D_ + 16 * j + coln;
      O[idx] = (__bf16)val;
    }
}

// ---------------------------------------------------------------------------
// Host orchestration
// ---------------------------------------------------------------------------
extern "C" void kernel_launch(void* const* d_in, const int* in_sizes, int n_in,
                              void* d_out, int out_size, void* d_ws, size_t ws_size,
                              hipStream_t stream) {
  const float* x  = (const float*)d_in[0];
  const float* fc = (const float*)d_in[1];
  const float* fs = (const float*)d_in[2];
  const float* wq = (const float*)d_in[3];
  const float* wk = (const float*)d_in[4];
  const float* wv = (const float*)d_in[5];
  const float* wo = (const float*)d_in[6];
  float* out = (float*)d_out;

  const size_t nX  = (size_t)B_ * S_ * DIM_;        // 8,388,608
  const size_t nWQ = (size_t)DIM_ * H_ * D_;        // 4,194,304
  const size_t nWK = (size_t)DIM_ * KVH_ * D_;      // 1,048,576
  const size_t nQ  = (size_t)B_ * S_ * H_ * D_;     // 8,388,608
  const size_t nK  = (size_t)B_ * S_ * KVH_ * D_;   // 2,097,152

  char* ws = (char*)d_ws;
  size_t off = 0;
  auto bfp = [&](size_t n) { __bf16* p = (__bf16*)(ws + off); off += n * sizeof(__bf16); return p; };
  auto fp  = [&](size_t n) { float*  p = (float*) (ws + off); off += n * sizeof(float);  return p; };

  __bf16* xb  = bfp(nX);      // x, row-major bf16
  __bf16* wqt = bfp(nWQ);     // weights transposed: [N][K]
  __bf16* wkt = bfp(nWK);
  __bf16* wvt = bfp(nWK);
  __bf16* wot = bfp(nWQ);
  float*  qf  = fp(nQ);
  float*  kf  = fp(nK);
  float*  vf  = fp(nK);
  __bf16* qrb = bfp(nQ);      // q after RoPE, bf16, (B,S,H,D)
  __bf16* krb = bfp(nK);      // k after RoPE, bf16, (B,S,KVH,D)
  __bf16* vtb = bfp(nK);      // v transposed, bf16, (B,KVH,D,S)
  __bf16* ob  = bfp(nQ);      // attention output, bf16, (B,S,H*D)

  const int T = 256;
  // 1) x -> bf16 (packed); weights -> bf16 transposed [N][K]
  f32_to_bf16x2_kernel<<<(nX / 2 + T - 1) / T, T, 0, stream>>>(
      (const float2*)x, (uint32_t*)xb, (int)(nX / 2));
  transpose_f32_to_bf16_kernel<<<dim3(DIM_ / 32, (H_ * D_) / 32),  128, 0, stream>>>(wq, wqt, DIM_, H_ * D_);
  transpose_f32_to_bf16_kernel<<<dim3(DIM_ / 32, (KVH_ * D_) / 32), 128, 0, stream>>>(wk, wkt, DIM_, KVH_ * D_);
  transpose_f32_to_bf16_kernel<<<dim3(DIM_ / 32, (KVH_ * D_) / 32), 128, 0, stream>>>(wv, wvt, DIM_, KVH_ * D_);
  transpose_f32_to_bf16_kernel<<<dim3((H_ * D_) / 32, DIM_ / 32),  128, 0, stream>>>(wo, wot, H_ * D_, DIM_);

  // 2) QKV projections (WMMA GEMMs), M = B*S = 4096
  const int M = B_ * S_;
  gemm_bf16_kernel<<<dim3((H_ * D_) / 64,   M / 64), 128, 0, stream>>>(xb, wqt, qf, M, H_ * D_,   DIM_);
  gemm_bf16_kernel<<<dim3((KVH_ * D_) / 64, M / 64), 128, 0, stream>>>(xb, wkt, kf, M, KVH_ * D_, DIM_);
  gemm_bf16_kernel<<<dim3((KVH_ * D_) / 64, M / 64), 128, 0, stream>>>(xb, wvt, vf, M, KVH_ * D_, DIM_);

  // 3) RoPE (q, k) -> bf16; transpose v -> (B,KVH,D,S) bf16
  const int totQ = (int)(nQ / 2), totK = (int)(nK / 2);
  rope_to_bf16_kernel<<<(totQ + T - 1) / T, T, 0, stream>>>(
      (const float2*)qf, fc, fs, (uint32_t*)qrb, H_, totQ);
  rope_to_bf16_kernel<<<(totK + T - 1) / T, T, 0, stream>>>(
      (const float2*)kf, fc, fs, (uint32_t*)krb, KVH_, totK);
  transpose_v_kernel<<<dim3(S_ / 32, D_ / 32, B_ * KVH_), 128, 0, stream>>>(vf, vtb);

  // 4) Causal GQA flash attention
  flash_attn_kernel<<<dim3(S_ / 64, H_, B_), 128, 0, stream>>>(qrb, krb, vtb, ob);

  // 5) Output projection -> f32 d_out
  gemm_bf16_kernel<<<dim3(DIM_ / 64, M / 64), 128, 0, stream>>>(ob, wot, out, M, DIM_, DIM_);
}
